// MahalanobisKnnModule_87093346828912
// MI455X (gfx1250) — matrix operand order repaired and checked by reading, not compiled, not run on hardware
//
#include <hip/hip_runtime.h>
#include <hip/hip_bf16.h>

#define NTRAIN 100000
#define DDIM   256
#define NEVAL  2048
#define NLAB   100
#define KNN    5

typedef __bf16 bf16;
typedef __bf16 v16bf __attribute__((ext_vector_type(16)));
typedef float  v8f   __attribute__((ext_vector_type(8)));

union Frag { v16bf v; uint4 q[2]; };

// ---- WMMA fragment loaders (16x16x32 bf16, wave32; contiguous k = 16*half + s, same for A and B) ----
// A (16x32, row-major source, ld elements): lane half=lane>>4, m=lane&15
//   VGPRs0-3: K = k0 + half*16 + 0..7    (16B contiguous)
//   VGPRs4-7: K = k0 + half*16 + 8..15   (16B contiguous, +16B)
__device__ __forceinline__ v16bf load_frag_A(const bf16* __restrict__ base,
                                             int row, int k0, int half, int ld) {
  Frag f;
  const char* p = (const char*)(base + (size_t)row * ld + k0 + half * 16);
  f.q[0] = *(const uint4*)p;
  f.q[1] = *(const uint4*)(p + 16);
  return f.v;
}
// B (32x16): lane m = N column; lanes0-15 hold K=k0..k0+15, lanes16-31 K=k0+16..k0+31.
// Source is "column rows contiguous in K" (X_eval row-major / M^T), so 32B contiguous.
__device__ __forceinline__ v16bf load_frag_B(const bf16* __restrict__ base,
                                             int col, int k0, int half, int ld) {
  Frag f;
  const uint4* p = (const uint4*)(base + (size_t)col * ld + k0 + half * 16);
  f.q[0] = p[0];
  f.q[1] = p[1];
  return f.v;
}

#define WMMA_BF16(A, B, C) \
  __builtin_amdgcn_wmma_f32_16x16x32_bf16(false, (A), false, (B), (short)0, (C), false, false)

// ---- conversions: f32 -> (hi, lo) bf16 split ----
__global__ void conv_hilo(const float* __restrict__ src, bf16* __restrict__ hi,
                          bf16* __restrict__ lo, size_t n) {
  size_t i = (size_t)blockIdx.x * blockDim.x + threadIdx.x;
  size_t stride = (size_t)gridDim.x * blockDim.x;
  for (; i < n; i += stride) {
    float x = src[i];
    bf16 h = (bf16)x;
    hi[i] = h;
    lo[i] = (bf16)(x - (float)h);
  }
}

// M[k][n] (256x256) -> MT[n][k] hi/lo
__global__ void conv_mt_hilo(const float* __restrict__ M, bf16* __restrict__ hiT,
                             bf16* __restrict__ loT) {
  int o = blockIdx.x * 256 + threadIdx.x;   // 65536 total
  int n = o >> 8, k = o & 255;
  float x = M[k * 256 + n];
  bf16 h = (bf16)x;
  hiT[o] = h;
  loT[o] = (bf16)(x - (float)h);
}

// ---- GEMM 1: cache = X_train @ M  -> stored as hi/lo bf16 ----
__global__ __launch_bounds__(256) void gemm_cache(
    const bf16* __restrict__ Ahi, const bf16* __restrict__ Alo,   // X_train hi/lo
    const bf16* __restrict__ Bhi, const bf16* __restrict__ Blo,   // M^T hi/lo
    bf16* __restrict__ Chi, bf16* __restrict__ Clo, int nTiles) {
  const int lane = threadIdx.x & 31;
  const int waveId = threadIdx.x >> 5;
  const int tileM = blockIdx.x * 8 + waveId;     // train tile (wave-uniform)
  if (tileM >= nTiles) return;
  const int tileN = blockIdx.y;                  // 0..15
  const int half = lane >> 4, m = lane & 15;
  const int rowA = tileM * 16 + m;
  const int colN = tileN * 16 + m;
  v8f acc = {0.f, 0.f, 0.f, 0.f, 0.f, 0.f, 0.f, 0.f};
  for (int k0 = 0; k0 < DDIM; k0 += 32) {
    v16bf ah = load_frag_A(Ahi, rowA, k0, half, DDIM);
    v16bf al = load_frag_A(Alo, rowA, k0, half, DDIM);
    v16bf bh = load_frag_B(Bhi, colN, k0, half, DDIM);
    v16bf bl = load_frag_B(Blo, colN, k0, half, DDIM);
    acc = WMMA_BF16(ah, bh, acc);
    acc = WMMA_BF16(ah, bl, acc);
    acc = WMMA_BF16(al, bh, acc);
  }
  // C layout: VGPR r -> row (half*8 + r), col m
#pragma unroll
  for (int r = 0; r < 8; ++r) {
    int tr = tileM * 16 + half * 8 + r;
    int cc = tileN * 16 + m;
    float x = acc[r];
    bf16 h = (bf16)x;
    Chi[(size_t)tr * DDIM + cc] = h;
    Clo[(size_t)tr * DDIM + cc] = (bf16)(x - (float)h);
  }
}

// ---- X_prod[i] = sum_k cache[i,k] * X_train[i,k]  (wave per row) ----
__global__ void xprod_kernel(const bf16* __restrict__ Chi, const bf16* __restrict__ Clo,
                             const float* __restrict__ Xt, float* __restrict__ Xp) {
  int gt = blockIdx.x * blockDim.x + threadIdx.x;
  int wave = gt >> 5, lane = gt & 31;
  int totalWaves = (gridDim.x * blockDim.x) >> 5;
  for (int i = wave; i < NTRAIN; i += totalWaves) {
    float s = 0.f;
    for (int k = lane; k < DDIM; k += 32) {
      size_t o = (size_t)i * DDIM + k;
      float c = (float)Chi[o] + (float)Clo[o];
      s += c * Xt[o];
    }
    for (int off = 16; off > 0; off >>= 1) s += __shfl_xor(s, off, 32);
    if (lane == 0) Xp[i] = s;
  }
}

// ---- Y_prod[j] = x_j^T M x_j  (block per eval row, pure f32) ----
__global__ __launch_bounds__(256) void yprod_kernel(const float* __restrict__ Xe,
                                                    const float* __restrict__ M,
                                                    float* __restrict__ Yp) {
  __shared__ float x[DDIM];
  __shared__ float red[256];
  int j = blockIdx.x;
  int d = threadIdx.x;
  x[d] = Xe[(size_t)j * DDIM + d];
  __syncthreads();
  float s = 0.f;
  for (int k = 0; k < DDIM; ++k) s += x[k] * M[k * DDIM + d];
  red[d] = s * x[d];
  __syncthreads();
  for (int off = 128; off > 0; off >>= 1) {
    if (d < off) red[d] += red[d + off];
    __syncthreads();
  }
  if (d == 0) Yp[j] = red[0];
}

// ---- GEMM 2 (dominant): csd[jLocal][i] = Xp[i] + Yp[j] - 2*(cache . x_eval_j) ----
__global__ __launch_bounds__(256) void gemm_csd(
    const bf16* __restrict__ Ahi, const bf16* __restrict__ Alo,   // cache hi/lo
    const bf16* __restrict__ Ehi, const bf16* __restrict__ Elo,   // X_eval hi/lo
    const float* __restrict__ Xp, const float* __restrict__ Yp,
    float* __restrict__ csd, int nTiles, int evalBase) {
  const int lane = threadIdx.x & 31;
  const int waveId = threadIdx.x >> 5;
  const int tileM = blockIdx.x * 8 + waveId;     // train tile (wave-uniform)
  if (tileM >= nTiles) return;
  const int tileN = blockIdx.y;                  // eval tile within chunk
  const int half = lane >> 4, m = lane & 15;
  const int rowA = tileM * 16 + m;
  const int evalRow = evalBase + tileN * 16 + m;
  v8f acc = {0.f, 0.f, 0.f, 0.f, 0.f, 0.f, 0.f, 0.f};
  for (int k0 = 0; k0 < DDIM; k0 += 32) {
    v16bf ah = load_frag_A(Ahi, rowA, k0, half, DDIM);
    v16bf al = load_frag_A(Alo, rowA, k0, half, DDIM);
    v16bf bh = load_frag_B(Ehi, evalRow, k0, half, DDIM);
    v16bf bl = load_frag_B(Elo, evalRow, k0, half, DDIM);
    acc = WMMA_BF16(ah, bh, acc);
    acc = WMMA_BF16(ah, bl, acc);
    acc = WMMA_BF16(al, bh, acc);
  }
  // epilogue: lane holds rows i = tileM*16 + half*8 + r, eval col j = evalRow
  const int ibase = tileM * 16 + half * 8;
  const float4* xp4 = (const float4*)(Xp + ibase);
  float4 xa = xp4[0], xb = xp4[1];
  float yp = Yp[evalRow];
  float4 o0, o1;
  o0.x = xa.x + yp - 2.f * acc[0];
  o0.y = xa.y + yp - 2.f * acc[1];
  o0.z = xa.z + yp - 2.f * acc[2];
  o0.w = xa.w + yp - 2.f * acc[3];
  o1.x = xb.x + yp - 2.f * acc[4];
  o1.y = xb.y + yp - 2.f * acc[5];
  o1.z = xb.z + yp - 2.f * acc[6];
  o1.w = xb.w + yp - 2.f * acc[7];
  float* dst = csd + (size_t)(tileN * 16 + m) * NTRAIN + ibase;
  ((float4*)dst)[0] = o0;
  ((float4*)dst)[1] = o1;
}

// ---- top-5 (smallest, ties -> lowest index) + label histogram ----
__global__ __launch_bounds__(256) void topk_label(const float* __restrict__ csd,
                                                  const int* __restrict__ ytrain,
                                                  float* __restrict__ out, int evalBase) {
  __shared__ float sv[256 * KNN];
  __shared__ int   si[256 * KNN];
  __shared__ int   lab[KNN];
  const int jLocal = blockIdx.x;
  const float* row = csd + (size_t)jLocal * NTRAIN;
  float bv[KNN] = {3.4e38f, 3.4e38f, 3.4e38f, 3.4e38f, 3.4e38f};
  int   bi[KNN] = {0x7fffffff, 0x7fffffff, 0x7fffffff, 0x7fffffff, 0x7fffffff};
#define TKSWAP(a, b) { float tv = bv[a]; bv[a] = bv[b]; bv[b] = tv; \
                       int ti = bi[a]; bi[a] = bi[b]; bi[b] = ti; }
  for (int i = threadIdx.x; i < NTRAIN; i += 256) {
    float d = row[i];
    if (d < bv[4]) {               // i ascends per-thread: ties keep lower index
      bv[4] = d; bi[4] = i;
      if (bv[4] < bv[3]) TKSWAP(4, 3);
      if (bv[3] < bv[2]) TKSWAP(3, 2);
      if (bv[2] < bv[1]) TKSWAP(2, 1);
      if (bv[1] < bv[0]) TKSWAP(1, 0);
    }
  }
#pragma unroll
  for (int t = 0; t < KNN; ++t) {
    sv[threadIdx.x * KNN + t] = bv[t];
    si[threadIdx.x * KNN + t] = bi[t];
  }
  __syncthreads();
  if (threadIdx.x == 0) {
    for (int r = 0; r < KNN; ++r) {
      float best = 3.4e38f;
      int bidx = 0x7fffffff, bslot = 0;
      for (int s = 0; s < 256 * KNN; ++s) {
        float v = sv[s];
        int ix = si[s];
        if (v < best || (v == best && ix < bidx)) { best = v; bidx = ix; bslot = s; }
      }
      sv[bslot] = 3.4e38f;
      lab[r] = (bidx >= 0 && bidx < NTRAIN) ? ytrain[bidx] : 0;
    }
  }
  __syncthreads();
  if (threadIdx.x < NLAB) {
    int c = 0;
#pragma unroll
    for (int r = 0; r < KNN; ++r) c += (lab[r] == (int)threadIdx.x);
    out[(size_t)(evalBase + jLocal) * NLAB + threadIdx.x] =
        (float)c - (float)threadIdx.x * (1.0f / NLAB);
  }
}

extern "C" void kernel_launch(void* const* d_in, const int* in_sizes, int n_in,
                              void* d_out, int out_size, void* d_ws, size_t ws_size,
                              hipStream_t stream) {
  const float* Xtr = (const float*)d_in[0];
  const float* M   = (const float*)d_in[1];
  const float* Xe  = (const float*)d_in[2];
  const int*   ytr = (const int*)d_in[3];
  float* out = (float*)d_out;

  char* ws = (char*)d_ws;
  size_t off = 0;
  auto take = [&](size_t bytes) -> char* {
    char* p = ws + off;
    off += (bytes + 255) & ~(size_t)255;
    return p;
  };
  bf16* XtHi = (bf16*)take((size_t)NTRAIN * DDIM * 2);
  bf16* XtLo = (bf16*)take((size_t)NTRAIN * DDIM * 2);
  bf16* CHi  = (bf16*)take((size_t)NTRAIN * DDIM * 2);
  bf16* CLo  = (bf16*)take((size_t)NTRAIN * DDIM * 2);
  bf16* MTHi = (bf16*)take((size_t)DDIM * DDIM * 2);
  bf16* MTLo = (bf16*)take((size_t)DDIM * DDIM * 2);
  bf16* XeHi = (bf16*)take((size_t)NEVAL * DDIM * 2);
  bf16* XeLo = (bf16*)take((size_t)NEVAL * DDIM * 2);
  float* Xp  = (float*)take((size_t)NTRAIN * 4);
  float* Yp  = (float*)take((size_t)NEVAL * 4);
  float* csd = (float*)(ws + off);

  // Eval chunk sized to fit the remaining workspace (csd = chunk * NTRAIN f32)
  size_t remain = (ws_size > off) ? (ws_size - off) : 0;
  size_t maxChunk = remain / ((size_t)NTRAIN * 4);
  int chunk = (int)(maxChunk > NEVAL ? NEVAL : maxChunk);
  chunk &= ~15;
  if (chunk < 16) chunk = 16;

  const int nTiles = NTRAIN / 16;                 // 6250
  const int gemmBlocksX = (nTiles + 7) / 8;       // 8 waves (tiles) per block

  conv_hilo<<<4096, 256, 0, stream>>>(Xtr, XtHi, XtLo, (size_t)NTRAIN * DDIM);
  conv_mt_hilo<<<256, 256, 0, stream>>>(M, MTHi, MTLo);
  conv_hilo<<<2048, 256, 0, stream>>>(Xe, XeHi, XeLo, (size_t)NEVAL * DDIM);

  gemm_cache<<<dim3(gemmBlocksX, DDIM / 16), 256, 0, stream>>>(
      XtHi, XtLo, MTHi, MTLo, CHi, CLo, nTiles);

  xprod_kernel<<<1024, 256, 0, stream>>>(CHi, CLo, Xtr, Xp);
  yprod_kernel<<<NEVAL, 256, 0, stream>>>(Xe, M, Yp);

  for (int base = 0; base < NEVAL; base += chunk) {
    int c = (NEVAL - base < chunk) ? (NEVAL - base) : chunk;
    gemm_csd<<<dim3(gemmBlocksX, c / 16), 256, 0, stream>>>(
        CHi, CLo, XeHi, XeLo, Xp, Yp, csd, nTiles, base);
    topk_label<<<c, 256, 0, stream>>>(csd, ytr, out, base);
  }
}
